// SinkhornSelfAttention_48747878809992
// MI455X (gfx1250) — matrix operand order repaired and checked by reading, not compiled, not run on hardware
//
#include <hip/hip_runtime.h>
#include <hip/hip_bf16.h>

// ---------------------------------------------------------------------------
// Sinkhorn self-attention for MI455X (gfx1250), wave32 + WMMA f16->f32.
// b=4 t=4096 d=1024 h=8 dh=128 bs=128 buckets=32
// ---------------------------------------------------------------------------

typedef _Float16 h16;
typedef __attribute__((ext_vector_type(16))) _Float16 v16h;
typedef __attribute__((ext_vector_type(8)))  _Float16 v8h;
typedef __attribute__((ext_vector_type(8)))  float    v8f;

union FragU { v16h v; v8h h[2]; };

// A-fragment (16x32 f16), this code's contiguous k order (B uses the same):
//   lane l: m = l&15 ; K = [kb, kb+16) contiguous, kb = 16*(l>>4)
// Source: row-major matrix, row stride ld (halves). Two 16B contiguous loads.
__device__ __forceinline__ v16h load_a_frag(const h16* base, int ld) {
  const int lane = threadIdx.x & 31;
  const int m  = lane & 15;
  const int kb = (lane >> 4) * 16;
  const h16* p = base + (size_t)m * ld + kb;
  FragU f;
  f.h[0] = *(const v8h*)(p);
  f.h[1] = *(const v8h*)(p + 8);
  return f.v;
}

// B-fragment (32x16 f16) where B = S^T and S is row-major [n][k] with row
// stride ld: lane l holds column n = l&15, K = 16*(l>>4) .. +15 (contiguous).
__device__ __forceinline__ v16h load_bt_frag(const h16* base, int ld) {
  const int lane = threadIdx.x & 31;
  const int n  = lane & 15;
  const int kb = (lane >> 4) * 16;
  const h16* p = base + (size_t)n * ld + kb;
  FragU f;
  f.h[0] = *(const v8h*)(p);
  f.h[1] = *(const v8h*)(p + 8);
  return f.v;
}

__device__ __forceinline__ v8f wmma16(v16h a, v16h b, v8f c) {
  return __builtin_amdgcn_wmma_f32_16x16x32_f16(false, a, false, b,
                                                (short)0, c, false, false);
}

// gfx1250 async global->LDS copy, 16 bytes per lane (tracked by ASYNCcnt).
__device__ __forceinline__ void async_copy_b128(unsigned lds_byte_off,
                                                const void* gaddr) {
  unsigned long long ga = (unsigned long long)(size_t)gaddr;
  asm volatile("global_load_async_to_lds_b128 %0, %1, off"
               :: "v"(lds_byte_off), "v"(ga) : "memory");
}
__device__ __forceinline__ void wait_asynccnt0() {
  asm volatile("s_wait_asynccnt 0x0" ::: "memory");
}

// ---------------------------------------------------------------------------
// fp32 -> f16 conversion, 8 elements/thread
// ---------------------------------------------------------------------------
__global__ void cvt_f32_to_f16(const float* __restrict__ in,
                               h16* __restrict__ out, int n8) {
  int i = blockIdx.x * blockDim.x + threadIdx.x;
  if (i >= n8) return;
  const float4* p = (const float4*)in + (size_t)i * 2;
  float4 a = p[0], b = p[1];
  v8h o = { (_Float16)a.x, (_Float16)a.y, (_Float16)a.z, (_Float16)a.w,
            (_Float16)b.x, (_Float16)b.y, (_Float16)b.z, (_Float16)b.w };
  ((v8h*)out)[i] = o;
}

// ---------------------------------------------------------------------------
// WMMA GEMM: C[M=16384][N] = A[16384][K=1024] (f16, lda) @ B[K][N] (f16 rowmaj)
// Block: 128x128 tile, 256 threads = 8 waves (4 M-groups x 2 N-groups).
// B slab staged transposed in LDS; A fragments loaded straight from global.
// FINAL: write f32 + bias to Cf, else f16 to Ch.
// ---------------------------------------------------------------------------
template <bool FINAL>
__global__ __launch_bounds__(256)
void gemm_wmma(const h16* __restrict__ A, int lda,
               const h16* __restrict__ B, int N,
               h16* __restrict__ Ch,
               float* __restrict__ Cf, const float* __restrict__ bias,
               int K)
{
  __shared__ h16 wt[128 * 40];   // B slab transposed: wt[n][k], ld=40 halves
  const int tid  = threadIdx.x;
  const int wave = tid >> 5;
  const int lane = tid & 31;
  const int wm = wave & 3;       // 32 rows per M-group
  const int wn = wave >> 2;      // 64 cols per N-group
  const int row0 = blockIdx.y * 128;
  const int col0 = blockIdx.x * 128;

  v8f acc[2][4];
  const v8f vzero = {};
#pragma unroll
  for (int i = 0; i < 2; ++i)
#pragma unroll
    for (int j = 0; j < 4; ++j) acc[i][j] = vzero;

  for (int k0 = 0; k0 < K; k0 += 32) {
    // cooperative transpose-load of B[k0..k0+31][col0..col0+127]
#pragma unroll
    for (int i = 0; i < 16; ++i) {
      int idx = i * 256 + tid;          // 0..4095
      int kk  = idx >> 7;               // 0..31
      int nn  = idx & 127;
      wt[nn * 40 + kk] = B[(size_t)(k0 + kk) * N + col0 + nn];
    }
    if (k0 + 32 < K)   // gfx1250 global_prefetch of next slab
      __builtin_prefetch(&B[(size_t)(k0 + 32 + (tid >> 7)) * N + col0 + (tid & 127)], 0, 1);
    __syncthreads();

    // cluster all fragment loads, then 8 independent WMMAs
    v16h a0 = load_a_frag(A + (size_t)(row0 + wm * 32 +  0) * lda + k0, lda);
    v16h a1 = load_a_frag(A + (size_t)(row0 + wm * 32 + 16) * lda + k0, lda);
    v16h bfr[4];
#pragma unroll
    for (int nt = 0; nt < 4; ++nt)
      bfr[nt] = load_bt_frag(&wt[(wn * 64 + nt * 16) * 40], 40);
#pragma unroll
    for (int nt = 0; nt < 4; ++nt) {
      acc[0][nt] = wmma16(a0, bfr[nt], acc[0][nt]);
      acc[1][nt] = wmma16(a1, bfr[nt], acc[1][nt]);
    }
    __syncthreads();
  }

  // C/D layout: m = (lane>>4)*8 + r, n = lane&15
#pragma unroll
  for (int mt = 0; mt < 2; ++mt)
#pragma unroll
    for (int nt = 0; nt < 4; ++nt) {
      int r0 = row0 + wm * 32 + mt * 16 + (lane >> 4) * 8;
      int c  = col0 + wn * 64 + nt * 16 + (lane & 15);
#pragma unroll
      for (int r = 0; r < 8; ++r) {
        float v = acc[mt][nt][r];
        if constexpr (FINAL) Cf[(size_t)(r0 + r) * N + c] = v + bias[c];
        else                 Ch[(size_t)(r0 + r) * N + c] = (h16)v;
      }
    }
}

// ---------------------------------------------------------------------------
// Bucket means: sq/sk[bh][u][dd] = mean over 128 rows.
// q_lin: [16384][1024], kv_lin: [16384][2048] (k = cols 0..1023)
// ---------------------------------------------------------------------------
__global__ void bucket_means(const h16* __restrict__ q_lin,
                             const h16* __restrict__ kv_lin,
                             float* __restrict__ sq, float* __restrict__ sk)
{
  int u = blockIdx.x, bh = blockIdx.y, dd = threadIdx.x;
  int b = bh >> 3, h = bh & 7;
  size_t qbase = (size_t)(b * 4096 + u * 128) * 1024 + h * 128 + dd;
  size_t kbase = (size_t)(b * 4096 + u * 128) * 2048 + h * 128 + dd;
  float s0 = 0.f, s1 = 0.f;
  for (int i = 0; i < 128; ++i) {
    s0 += (float)q_lin[qbase + (size_t)i * 1024];
    s1 += (float)kv_lin[kbase + (size_t)i * 2048];
  }
  sq[(bh * 32 + u) * 128 + dd] = s0 * (1.f / 128.f);
  sk[(bh * 32 + u) * 128 + dd] = s1 * (1.f / 128.f);
}

// ---------------------------------------------------------------------------
// Sort net: R[bh][u][v] = softmax_v(sq[u].sk[v] * dh^-0.5). 32 blocks x 32 thr
// ---------------------------------------------------------------------------
__global__ void sort_net(const float* __restrict__ sq,
                         const float* __restrict__ sk, float* __restrict__ R)
{
  int bh = blockIdx.x, u = threadIdx.x;
  const float scale = 0.08838834764831845f;   // 128^-0.5
  float row[32];
  float m = -1e30f;
  for (int v = 0; v < 32; ++v) {
    float d = 0.f;
    for (int e = 0; e < 128; ++e)
      d += sq[(bh * 32 + u) * 128 + e] * sk[(bh * 32 + v) * 128 + e];
    d *= scale;
    row[v] = d;
    m = fmaxf(m, d);
  }
  float s = 0.f;
  for (int v = 0; v < 32; ++v) { row[v] = __expf(row[v] - m); s += row[v]; }
  float inv = 1.f / s;
  for (int v = 0; v < 32; ++v) R[(bh * 32 + u) * 32 + v] = row[v] * inv;
}

// ---------------------------------------------------------------------------
// Soft reorder: out[bh][u][i][dd] = sum_v R[bh][u][v] * src[bh][v][i][dd]
// src element read from kv_lin with merged-head indexing (voff 0 = K, 1024 = V)
// grid (64 chunks, 32 bh) x 256 thr, 1 (i,dd) pair per thread, 32 outputs.
// ---------------------------------------------------------------------------
__global__ __launch_bounds__(256)
void soft_reorder(const float* __restrict__ R, const h16* __restrict__ kv_lin,
                  int voff, h16* __restrict__ out)
{
  __shared__ float Rs[1024];
  int bh = blockIdx.y, b = bh >> 3, h = bh & 7;
  int tid = threadIdx.x;
  for (int i = tid; i < 1024; i += 256) Rs[i] = R[bh * 1024 + i];
  __syncthreads();

  int p  = blockIdx.x * 256 + tid;   // 0..16383
  int i  = p >> 7, dd = p & 127;
  float acc[32];
#pragma unroll
  for (int u = 0; u < 32; ++u) acc[u] = 0.f;
  for (int v = 0; v < 32; ++v) {
    float kv = (float)kv_lin[(size_t)(b * 4096 + v * 128 + i) * 2048
                             + voff + h * 128 + dd];
#pragma unroll
    for (int u = 0; u < 32; ++u) acc[u] += Rs[u * 32 + v] * kv;
  }
  for (int u = 0; u < 32; ++u)
    out[((size_t)(bh * 32 + u) * 128 + i) * 128 + dd] = (h16)acc[u];
}

// ---------------------------------------------------------------------------
// Per-bucket attention, one block per (u, bh). 256 thr = 8 waves.
//  - async-prefetch local V bucket into LDS (ASYNCcnt) at kernel start
//  pass1: dots[128][256] = scale * q @ [kr ; k]^T  (WMMA, B direct from global,
//         two key tiles in flight -> independent WMMA chains)
//  softmax rows (f32, in LDS, 2 threads/row) -> attn f16
//  pass2: out = attn @ [vr ; v]  (WMMA, A from LDS attn, B from LDS v^T)
// Dynamic LDS: dots 128K + attn 64K + vT 64K + vbuf 32K = 288 KB (<320 KB/WG)
// ---------------------------------------------------------------------------
__global__ __launch_bounds__(256)
void bucket_attention(const h16* __restrict__ q_lin,
                      const h16* __restrict__ kv_lin,
                      const h16* __restrict__ kr, const h16* __restrict__ vr,
                      h16* __restrict__ oh)
{
  extern __shared__ char smem[];
  float* dots = (float*)smem;                                   // [128][256]
  h16*   attn = (h16*)(smem + 131072);                          // [128][256]
  h16*   vT   = (h16*)(smem + 131072 + 65536);                  // [dh][256 key]
  h16*   vbuf = (h16*)(smem + 131072 + 65536 + 65536);          // [128][128]

  const int u = blockIdx.x, bh = blockIdx.y;
  const int b = bh >> 3, h = bh & 7;
  const int tid = threadIdx.x, wave = tid >> 5, lane = tid & 31;
  const float scale = 0.08838834764831845f;

  const h16* qb  = q_lin  + (size_t)(b * 4096 + u * 128) * 1024 + h * 128;
  const h16* kb  = kv_lin + (size_t)(b * 4096 + u * 128) * 2048 + h * 128;
  const h16* vb  = kv_lin + (size_t)(b * 4096 + u * 128) * 2048 + 1024 + h * 128;
  const h16* krb = kr + (size_t)(bh * 32 + u) * 128 * 128;
  const h16* vrb = vr + (size_t)(bh * 32 + u) * 128 * 128;

  // ---- async prefetch of local V bucket into vbuf (overlaps pass 1) ------
  // 128 rows x 256B; 2048 chunks of 16B; 8 chunks per thread.
  const unsigned vbuf_off = (unsigned)(size_t)(const void*)vbuf;
#pragma unroll
  for (int c8 = 0; c8 < 8; ++c8) {
    int c   = c8 * 256 + tid;
    int row = c >> 4, seg = c & 15;
    async_copy_b128(vbuf_off + (unsigned)(row * 256 + seg * 16),
                    vb + (size_t)row * 2048 + seg * 8);
  }

  // ---- pass 1: wave w owns query rows [w*16, w*16+16) --------------------
  v16h aq[4];
#pragma unroll
  for (int ks = 0; ks < 4; ++ks)
    aq[ks] = load_a_frag(qb + (size_t)(wave * 16) * 1024 + ks * 32, 1024);

  const v8f vzero = {};
#pragma unroll 1
  for (int nt = 0; nt < 16; nt += 2) {     // 2 key tiles per iter (0..7 = kr)
    const h16* base; int bld, idx;
    if (nt < 8) { base = krb; bld = 128;  idx = nt;     }
    else        { base = kb;  bld = 2048; idx = nt - 8; }
    const h16* bsrc0 = base + (size_t)(idx * 16)       * bld;
    const h16* bsrc1 = base + (size_t)((idx + 1) * 16) * bld;
    v16h bf0[4], bf1[4];
#pragma unroll
    for (int ks = 0; ks < 4; ++ks)         // 16 clustered global_load_b128
      bf0[ks] = load_bt_frag(bsrc0 + ks * 32, bld);
#pragma unroll
    for (int ks = 0; ks < 4; ++ks)
      bf1[ks] = load_bt_frag(bsrc1 + ks * 32, bld);
    v8f acc0 = vzero, acc1 = vzero;        // two independent WMMA chains
#pragma unroll
    for (int ks = 0; ks < 4; ++ks) {
      acc0 = wmma16(aq[ks], bf0[ks], acc0);
      acc1 = wmma16(aq[ks], bf1[ks], acc1);
    }
    acc0 *= scale;
    acc1 *= scale;
    int m0 = wave * 16 + (lane >> 4) * 8;
    int n0 = nt * 16 + (lane & 15);
#pragma unroll
    for (int r = 0; r < 8; ++r) {
      dots[(m0 + r) * 256 + n0]      = acc0[r];
      dots[(m0 + r) * 256 + n0 + 16] = acc1[r];
    }
  }
  wait_asynccnt0();           // V bucket resident in vbuf after the barrier
  __syncthreads();

  // ---- v^T staging (all threads; vr from global, local v from vbuf) ------
  for (int e = tid; e < 128 * 256; e += 256) {
    int dd = e & 127, key = e >> 7;
    h16 val = (key < 128) ? vrb[(size_t)key * 128 + dd]
                          : vbuf[(key - 128) * 128 + dd];
    vT[dd * 256 + key] = val;
  }
  // ---- row softmax: 2 threads per row, wave32 shuffle reduction ----------
  {
    int row = tid >> 1, half = tid & 1;
    int j0 = half * 128, j1 = j0 + 128;
    float m = -1e30f;
    for (int j = j0; j < j1; ++j) m = fmaxf(m, dots[row * 256 + j]);
    m = fmaxf(m, __shfl_xor(m, 1, 32));
    float s = 0.f;
    for (int j = j0; j < j1; ++j) {
      float ev = __expf(dots[row * 256 + j] - m);
      s += ev;
      dots[row * 256 + j] = ev;
    }
    s += __shfl_xor(s, 1, 32);
    float inv = 1.f / s;
    for (int j = j0; j < j1; ++j)
      attn[row * 256 + j] = (h16)(dots[row * 256 + j] * inv);
  }
  __syncthreads();

  // ---- pass 2: out[128][128] = attn @ v_cat ------------------------------
  v8f acc2[8];
#pragma unroll
  for (int nt = 0; nt < 8; ++nt) acc2[nt] = vzero;
#pragma unroll 1
  for (int ks = 0; ks < 8; ++ks) {         // K = 256 keys
    v16h af = load_a_frag(attn + (wave * 16) * 256 + ks * 32, 256);
#pragma unroll
    for (int half = 0; half < 2; ++half) {
      v16h bfr[4];
#pragma unroll
      for (int j = 0; j < 4; ++j)          // clause of 8 ds_load_b128
        bfr[j] = load_bt_frag(vT + ((half * 4 + j) * 16) * 256 + ks * 32, 256);
#pragma unroll
      for (int j = 0; j < 4; ++j)          // 4 independent WMMAs
        acc2[half * 4 + j] = wmma16(af, bfr[j], acc2[half * 4 + j]);
    }
  }
  // write merged-head layout: row (b*4096 + u*128 + ...), col h*128 + ...
  h16* ob = oh + (size_t)(b * 4096 + u * 128 + wave * 16) * 1024 + h * 128;
#pragma unroll
  for (int nt = 0; nt < 8; ++nt) {
    int m0 = (lane >> 4) * 8;
    int n  = nt * 16 + (lane & 15);
#pragma unroll
    for (int r = 0; r < 8; ++r)
      ob[(size_t)(m0 + r) * 1024 + n] = (h16)acc2[nt][r];
  }
}

// ---------------------------------------------------------------------------
// Launch
// ---------------------------------------------------------------------------
extern "C" void kernel_launch(void* const* d_in, const int* in_sizes, int n_in,
                              void* d_out, int out_size, void* d_ws, size_t ws_size,
                              hipStream_t stream)
{
  (void)in_sizes; (void)n_in; (void)out_size; (void)ws_size;
  const float* x   = (const float*)d_in[0];
  const float* Wq  = (const float*)d_in[1];
  const float* Wkv = (const float*)d_in[2];
  const float* Wo  = (const float*)d_in[3];
  const float* bo  = (const float*)d_in[4];
  float* out = (float*)d_out;

  char* ws = (char*)d_ws;
  size_t off = 0;
  auto alloc = [&](size_t bytes) -> char* {
    char* p = ws + off; off += (bytes + 255) & ~(size_t)255; return p;
  };
  h16*   xh    = (h16*)alloc(16384ull * 1024 * 2);
  h16*   Wqh   = (h16*)alloc(1024ull * 1024 * 2);
  h16*   Wkvh  = (h16*)alloc(1024ull * 2048 * 2);
  h16*   Woh   = (h16*)alloc(1024ull * 1024 * 2);
  h16*   qlin  = (h16*)alloc(16384ull * 1024 * 2);
  h16*   kvlin = (h16*)alloc(16384ull * 2048 * 2);
  float* sq    = (float*)alloc(32ull * 32 * 128 * 4);
  float* sk    = (float*)alloc(32ull * 32 * 128 * 4);
  float* R     = (float*)alloc(32ull * 32 * 32 * 4);
  h16*   kr    = (h16*)alloc(32ull * 32 * 128 * 128 * 2);
  h16*   vr    = (h16*)alloc(32ull * 32 * 128 * 128 * 2);
  h16*   oh    = (h16*)alloc(16384ull * 1024 * 2);

  // fp32 -> f16 staging
  cvt_f32_to_f16<<<(16384 * 1024 / 8 + 255) / 256, 256, 0, stream>>>(x,   xh,   16384 * 1024 / 8);
  cvt_f32_to_f16<<<(1024 * 1024 / 8 + 255) / 256, 256, 0, stream>>>(Wq,  Wqh,  1024 * 1024 / 8);
  cvt_f32_to_f16<<<(1024 * 2048 / 8 + 255) / 256, 256, 0, stream>>>(Wkv, Wkvh, 1024 * 2048 / 8);
  cvt_f32_to_f16<<<(1024 * 1024 / 8 + 255) / 256, 256, 0, stream>>>(Wo,  Woh,  1024 * 1024 / 8);

  // projections (WMMA)
  gemm_wmma<false><<<dim3(1024 / 128, 16384 / 128), 256, 0, stream>>>(
      xh, 1024, Wqh, 1024, qlin, nullptr, nullptr, 1024);
  gemm_wmma<false><<<dim3(2048 / 128, 16384 / 128), 256, 0, stream>>>(
      xh, 1024, Wkvh, 2048, kvlin, nullptr, nullptr, 1024);

  // sort net + soft bucket reorder
  bucket_means<<<dim3(32, 32), 128, 0, stream>>>(qlin, kvlin, sq, sk);
  sort_net<<<32, 32, 0, stream>>>(sq, sk, R);
  soft_reorder<<<dim3(64, 32), 256, 0, stream>>>(R, kvlin, 0,    kr);
  soft_reorder<<<dim3(64, 32), 256, 0, stream>>>(R, kvlin, 1024, vr);

  // per-bucket attention (WMMA + async LDS prefetch), 288 KB dynamic LDS
  (void)hipFuncSetAttribute((const void*)bucket_attention,
                            hipFuncAttributeMaxDynamicSharedMemorySize, 294912);
  bucket_attention<<<dim3(32, 32), 256, 294912, stream>>>(qlin, kvlin, kr, vr, oh);

  // output projection with bias (WMMA, f32 out)
  gemm_wmma<true><<<dim3(1024 / 128, 16384 / 128), 256, 0, stream>>>(
      oh, 1024, Woh, 1024, nullptr, out, bo, 1024);
}
